// Critic_26147760898237
// MI455X (gfx1250) — matrix operand order repaired and checked by reading, not compiled, not run on hardware
//
#include <hip/hip_runtime.h>

typedef __bf16 bf16;
typedef __attribute__((ext_vector_type(16))) __bf16 v16bf;
typedef __attribute__((ext_vector_type(8)))  float  v8f;

#define EPS_BN 1e-3f

union U16 { uint4 u; bf16 h[8]; };
union U4  { unsigned u; bf16 h[2]; };

// ---------------------------------------------------------------------------
// Input prep: pack channel0 (bf16) + zeroed channel1 into one b32 store.
// act0 layout: [B=64][256][256][C=2] bf16
// ---------------------------------------------------------------------------
__global__ void prep_input_kernel(const float* __restrict__ x, bf16* __restrict__ act0) {
  int idx = blockIdx.x * blockDim.x + threadIdx.x;     // over 64*65536 pixels
  if (idx >= 64 * 65536) return;
  int b = idx >> 16;
  int p = idx & 65535;
  float v = x[(size_t)(b * 2) * 65536 + p];            // channel 0
  U4 o; o.h[0] = (bf16)v; o.h[1] = (bf16)0.0f;
  ((unsigned*)act0)[idx] = o.u;
}

// ---------------------------------------------------------------------------
// Rasterize the segment between the two pixels == 2.0 in channel 1.
// One block of 256 threads per batch image.
// ---------------------------------------------------------------------------
__global__ void rasterize_kernel(const float* __restrict__ x, bf16* __restrict__ act0) {
  int b = blockIdx.x;
  const float* ch1 = x + ((size_t)(b * 2 + 1)) * 65536;
  __shared__ unsigned s1, s2;
  if (threadIdx.x == 0) { s1 = 0xFFFFFFFFu; s2 = 0xFFFFFFFFu; }
  __syncthreads();
  for (int i = threadIdx.x; i < 65536; i += blockDim.x)
    if (ch1[i] == 2.0f) atomicMin(&s1, (unsigned)i);
  __syncthreads();
  for (int i = threadIdx.x; i < 65536; i += blockDim.x)
    if (ch1[i] == 2.0f && (unsigned)i != s1) atomicMin(&s2, (unsigned)i);
  __syncthreads();
  int i0 = (int)s1, i1 = (int)s2;                      // ascending row-major order
  int y1 = i0 >> 8, x1 = i0 & 255;
  int y2 = i1 >> 8, x2 = i1 & 255;
  int dx = x2 - x1, dy = y2 - y1;
  bf16* e = act0 + ((size_t)b) * 65536 * 2;
  int i = threadIdx.x;                                  // 0..255 == 'ar'
  if (i < 256) {
    bool mx = (x1 < x2) ? (i >= x1 && i < x2) : ((x1 > x2) ? (i > x2 && i <= x1) : false);
    bool my = (y1 < y2) ? (i >= y1 && i < y2) : ((y1 > y2) ? (i > y2 && i <= y1) : false);
    if (mx) {
      float m1 = (float)dy / (float)(dx == 0 ? 1 : dx);
      int ys = (int)rintf(m1 * (float)(i - x1) + (float)y1);   // RNE like jnp.round
      ys = min(max(ys, 0), 255);
      e[((size_t)(ys * 256 + i)) * 2 + 1] = (bf16)1.0f;
    }
    if (my) {
      float m2 = (float)dx / (float)(dy == 0 ? 1 : dy);
      int xs = (int)rintf(m2 * (float)(i - y1) + (float)x1);
      xs = min(max(xs, 0), 255);
      e[((size_t)(i * 256 + xs)) * 2 + 1] = (bf16)1.0f;
    }
  }
  __syncthreads();
  if (threadIdx.x == 0) {                               // endpoints written last
    e[((size_t)(y1 * 256 + x1)) * 2 + 1] = (bf16)2.0f;
    e[((size_t)(y2 * 256 + x2)) * 2 + 1] = (bf16)2.0f;
  }
}

// ---------------------------------------------------------------------------
// Convert OIHW fp32 weights into a bf16 GEMM matrix Wmat[C_out_pad][K],
// K = 16*C_in with k = (ky*4+kx)*C_in + ci (tap-major, channel-minor).
// ---------------------------------------------------------------------------
__global__ void prep_weights_kernel(const float* __restrict__ w, bf16* __restrict__ wm,
                                    int C_in, int C_out, int C_out_pad, int cin_shift) {
  int k_shift = cin_shift + 4;                          // K = C_in*16 (power of 2)
  int K = C_in << 4;
  int total = C_out_pad * K;
  for (int idx = blockIdx.x * blockDim.x + threadIdx.x; idx < total;
       idx += gridDim.x * blockDim.x) {
    int cout = idx >> k_shift;
    int k = idx & (K - 1);
    int ci = k & (C_in - 1);
    int tap = k >> cin_shift;
    int ky = tap >> 2, kx = tap & 3;
    float v = (cout < C_out) ? w[(((size_t)cout * C_in + ci) * 4 + ky) * 4 + kx] : 0.0f;
    wm[idx] = (bf16)v;
  }
}

// ---------------------------------------------------------------------------
// Fragment loaders (same contiguous k order for A and B), b128-vectorized.
// A 16x32 bf16: lane row = cout_base+(lane&15); K=khalf*16+j
//               => two contiguous 16B groups.
// B 32x16 bf16: N=lane&15, K=khalf*16+j     => one contiguous 32B group.
// ---------------------------------------------------------------------------
__device__ __forceinline__ v16bf load_a_b128(const bf16* __restrict__ wrow, int khalf) {
  U16 lo, hi;
  lo.u = *(const uint4*)(wrow + (khalf << 4));
  hi.u = *(const uint4*)(wrow + 8 + (khalf << 4));
  v16bf a;
  #pragma unroll
  for (int j = 0; j < 8; ++j) { a[j] = lo.h[j]; a[j + 8] = hi.h[j]; }
  return a;
}

__device__ __forceinline__ v16bf load_b_b128(const bf16* __restrict__ p, bool valid) {
  v16bf f;
  if (valid) {
    U16 lo, hi;
    lo.u = *(const uint4*)(p);
    hi.u = *(const uint4*)(p + 8);
    #pragma unroll
    for (int j = 0; j < 8; ++j) { f[j] = lo.h[j]; f[j + 8] = hi.h[j]; }
  } else {
    #pragma unroll
    for (int j = 0; j < 16; ++j) f[j] = (bf16)0.0f;
  }
  return f;
}

// ---------------------------------------------------------------------------
// Implicit-GEMM conv: D[cout][pixel] = sum_k Wmat[cout][k] * im2col[k][pixel].
// One wave computes a 16(cout) x 64(pixel) tile: 4 WMMA accumulators reusing
// one A fragment per K-chunk of 32; K-loop unrolled x2 (8 WMMAs/iteration).
// Vector path (C_in>=64): all A/B traffic as global_load_b128 with a single
// bounds check per 16-channel group; weight panel prefetched ahead.
// flags: bit0 = LeakyReLU in conv (layer 0); bit1 = write fp32 (layer 5).
// ---------------------------------------------------------------------------
__global__ void __launch_bounds__(128)
conv_wmma_kernel(const bf16* __restrict__ in, const bf16* __restrict__ wmat,
                 const float* __restrict__ bias,
                 bf16* __restrict__ out_bf, float* __restrict__ out_f32,
                 int C_in, int cin_shift, int C_out,
                 int H_in, int W_in, int W_out, int hw_shift,
                 int stride, int total_pix, int flags) {
  const int lane  = threadIdx.x & 31;
  const int wave  = threadIdx.x >> 5;
  const int khalf = lane >> 4;                          // 0 or 1 (half-wave)
  const int K_total = C_in << 4;
  const int cout_base = blockIdx.y << 4;
  const int nbase = (blockIdx.x << 8) + (wave << 6);    // 64 pixels per wave

  // Decode the 4 pixel groups (16 pixels each) handled by this wave.
  int pb[4], pys[4], pxs[4]; bool pv[4];
  const int wmask = W_out - 1;
  #pragma unroll
  for (int t = 0; t < 4; ++t) {
    int n = nbase + (t << 4) + (lane & 15);
    pv[t] = (n < total_pix);
    int nn = pv[t] ? n : 0;
    pxs[t] = (nn & wmask) * stride - 1;                 // pad_low = 1 every layer
    int r = nn >> hw_shift;
    pys[t] = (r & wmask) * stride - 1;                  // H_out == W_out
    pb[t] = r >> hw_shift;
  }

  v8f acc0 = {}, acc1 = {}, acc2 = {}, acc3 = {};
  const bf16* wrow = wmat + (size_t)(cout_base + (lane & 15)) * K_total;

  if (C_in >= 64) {
    // ---- vectorized path: 16 K-values per lane = 16 consecutive channels ----
    const int cmask = C_in - 1;
    for (int kc = 0; kc < K_total; kc += 64) {
      if (kc + 64 < K_total)
        __builtin_prefetch(wrow + kc + 64, 0, 1);       // global_prefetch_b8
      #pragma unroll
      for (int h = 0; h < 2; ++h) {
        const int kcc = kc + (h << 5);
        v16bf a = load_a_b128(wrow + kcc, khalf);
        const int kb  = kcc + (khalf << 4);
        const int ci  = kb & cmask;                     // 16-aligned channel base
        const int tap = kb >> cin_shift;                // constant over the group
        const int dyt = tap >> 2, dxt = tap & 3;
        v16bf b0, b1, b2, b3;
        #pragma unroll
        for (int t = 0; t < 4; ++t) {
          int iy = pys[t] + dyt, ix = pxs[t] + dxt;
          bool ok = pv[t] && iy >= 0 && iy < H_in && ix >= 0 && ix < W_in;
          const bf16* p = in + (((size_t)pb[t] * H_in + iy) * W_in + ix) * C_in + ci;
          v16bf f = load_b_b128(p, ok);
          if (t == 0) b0 = f; else if (t == 1) b1 = f; else if (t == 2) b2 = f; else b3 = f;
        }
        acc0 = __builtin_amdgcn_wmma_f32_16x16x32_bf16(false, a, false, b0, (short)0, acc0, false, false);
        acc1 = __builtin_amdgcn_wmma_f32_16x16x32_bf16(false, a, false, b1, (short)0, acc1, false, false);
        acc2 = __builtin_amdgcn_wmma_f32_16x16x32_bf16(false, a, false, b2, (short)0, acc2, false, false);
        acc3 = __builtin_amdgcn_wmma_f32_16x16x32_bf16(false, a, false, b3, (short)0, acc3, false, false);
      }
    }
  } else {
    // ---- scalar gather path (layer 0: C_in=2, K_total=32, one chunk) ----
    for (int kc = 0; kc < K_total; kc += 32) {
      v16bf a;
      {
        const bf16* wp = wrow + kc;
        #pragma unroll
        for (int j = 0; j < 16; ++j) {
          int k = j + (khalf << 4);
          a[j] = wp[k];
        }
      }
      v16bf b0, b1, b2, b3;
      #pragma unroll
      for (int t = 0; t < 4; ++t) {
        v16bf f;
        #pragma unroll
        for (int j = 0; j < 16; ++j) {
          int k   = kc + (khalf << 4) + j;
          int ci  = k & (C_in - 1);
          int tap = k >> cin_shift;
          int iy  = pys[t] + (tap >> 2);
          int ix  = pxs[t] + (tap & 3);
          bf16 v = (bf16)0.0f;
          if (pv[t] && iy >= 0 && iy < H_in && ix >= 0 && ix < W_in)
            v = in[(((size_t)pb[t] * H_in + iy) * W_in + ix) * C_in + ci];
          f[j] = v;
        }
        if (t == 0) b0 = f; else if (t == 1) b1 = f; else if (t == 2) b2 = f; else b3 = f;
      }
      acc0 = __builtin_amdgcn_wmma_f32_16x16x32_bf16(false, a, false, b0, (short)0, acc0, false, false);
      acc1 = __builtin_amdgcn_wmma_f32_16x16x32_bf16(false, a, false, b1, (short)0, acc1, false, false);
      acc2 = __builtin_amdgcn_wmma_f32_16x16x32_bf16(false, a, false, b2, (short)0, acc2, false, false);
      acc3 = __builtin_amdgcn_wmma_f32_16x16x32_bf16(false, a, false, b3, (short)0, acc3, false, false);
    }
  }

  // Store: C/D layout — VGPR r holds M = r + 8*khalf, N = lane&15.
  // The 8 rows per lane are 8 consecutive output channels -> one b128 store.
  v8f accs[4] = {acc0, acc1, acc2, acc3};
  if (flags & 2) {
    // layer 5: only cout==0 meaningful, fp32 to d_out
    if (cout_base == 0 && khalf == 0) {
      float bv = bias[0];
      #pragma unroll
      for (int t = 0; t < 4; ++t) {
        int n_col = nbase + (t << 4) + (lane & 15);
        if (n_col < total_pix) out_f32[n_col] = accs[t][0] + bv;
      }
    }
  } else {
    const int cbase = cout_base + (khalf << 3);
    float bset[8];
    #pragma unroll
    for (int r = 0; r < 8; ++r) bset[r] = bias[cbase + r];
    #pragma unroll
    for (int t = 0; t < 4; ++t) {
      int n_col = nbase + (t << 4) + (lane & 15);
      if (n_col >= total_pix) continue;
      U16 o;
      #pragma unroll
      for (int r = 0; r < 8; ++r) {
        float v = accs[t][r] + bset[r];
        if (flags & 1) v = (v >= 0.0f) ? v : 0.2f * v;  // LeakyReLU (layer 0)
        o.h[r] = (bf16)v;
      }
      *(uint4*)(out_bf + (size_t)n_col * C_out + cbase) = o.u;
    }
  }
}

// ---------------------------------------------------------------------------
// BatchNorm pass 1: per-channel sum / sumsq via LDS float atomics.
// Vectorized x8: each b128 load covers 8 consecutive channels.
// sums[c] = sum, sums[C+c] = sumsq. C is a power of 2 (64..512).
// ---------------------------------------------------------------------------
__global__ void stats_kernel(const bf16* __restrict__ act, float* __restrict__ sums,
                             int C, size_t total8) {
  __shared__ float s[1024];                             // 2*512 max
  int twoC = 2 * C;
  for (int t = threadIdx.x; t < twoC; t += blockDim.x) s[t] = 0.0f;
  __syncthreads();
  size_t cmask = (size_t)(C - 1);
  for (size_t idx = (size_t)blockIdx.x * blockDim.x + threadIdx.x; idx < total8;
       idx += (size_t)gridDim.x * blockDim.x) {
    int c0 = (int)((idx * 8) & cmask);
    U16 v; v.u = ((const uint4*)act)[idx];
    #pragma unroll
    for (int j = 0; j < 8; ++j) {
      float f = (float)v.h[j];
      atomicAdd(&s[c0 + j], f);
      atomicAdd(&s[C + c0 + j], f * f);
    }
  }
  __syncthreads();
  for (int t = threadIdx.x; t < twoC; t += blockDim.x)
    atomicAdd(&sums[t], s[t]);
}

__global__ void zero_kernel(float* __restrict__ p, int n) {
  int i = blockIdx.x * blockDim.x + threadIdx.x;
  if (i < n) p[i] = 0.0f;
}

// scale = g*rsqrt(var+eps); shift = be - mean*scale
__global__ void bn_finalize_kernel(const float* __restrict__ sums, const float* __restrict__ g,
                                   const float* __restrict__ be, float* __restrict__ scsh,
                                   int C, float inv_n) {
  int c = blockIdx.x * blockDim.x + threadIdx.x;
  if (c >= C) return;
  float mean = sums[c] * inv_n;
  float var  = sums[C + c] * inv_n - mean * mean;
  float sc = g[c] * rsqrtf(var + EPS_BN);
  scsh[c]     = sc;
  scsh[C + c] = be[c] - mean * sc;
}

// BatchNorm pass 2: normalize + LeakyReLU, in place, b128 load/store.
__global__ void bn_apply_kernel(bf16* __restrict__ act, const float* __restrict__ scsh,
                                int C, size_t total8) {
  size_t cmask = (size_t)(C - 1);
  for (size_t idx = (size_t)blockIdx.x * blockDim.x + threadIdx.x; idx < total8;
       idx += (size_t)gridDim.x * blockDim.x) {
    int c0 = (int)((idx * 8) & cmask);
    U16 v; v.u = ((const uint4*)act)[idx];
    #pragma unroll
    for (int j = 0; j < 8; ++j) {
      float f = (float)v.h[j] * scsh[c0 + j] + scsh[C + c0 + j];
      f = (f >= 0.0f) ? f : 0.2f * f;
      v.h[j] = (bf16)f;
    }
    ((uint4*)act)[idx] = v.u;
  }
}

// ---------------------------------------------------------------------------
extern "C" void kernel_launch(void* const* d_in, const int* in_sizes, int n_in,
                              void* d_out, int out_size, void* d_ws, size_t ws_size,
                              hipStream_t stream) {
  (void)in_sizes; (void)n_in; (void)out_size; (void)ws_size;

  const float* x = (const float*)d_in[0];
  const float* w[6]  = {(const float*)d_in[1], (const float*)d_in[3],  (const float*)d_in[5],
                        (const float*)d_in[7], (const float*)d_in[9],  (const float*)d_in[11]};
  const float* bb[6] = {(const float*)d_in[2], (const float*)d_in[4],  (const float*)d_in[6],
                        (const float*)d_in[8], (const float*)d_in[10], (const float*)d_in[12]};
  const float* g[6]  = {nullptr, (const float*)d_in[13], (const float*)d_in[15],
                        (const float*)d_in[17], (const float*)d_in[19], nullptr};
  const float* be[6] = {nullptr, (const float*)d_in[14], (const float*)d_in[16],
                        (const float*)d_in[18], (const float*)d_in[20], nullptr};

  // Workspace layout (bf16 NHWC activations + bf16 GEMM weights + BN scratch)
  char* ws = (char*)d_ws;
  size_t off = 0;
  auto alloc = [&](size_t bytes) -> char* {
    char* p = ws + off;
    off = (off + bytes + 255) & ~(size_t)255;
    return p;
  };
  const size_t actElems[6] = {8388608ull, 67108864ull, 33554432ull,
                              16777216ull, 8388608ull, 8388608ull};
  bf16* act[6];
  for (int i = 0; i < 6; ++i) act[i] = (bf16*)alloc(actElems[i] * 2);
  const size_t wElems[6] = {2048ull, 131072ull, 524288ull, 2097152ull, 4194304ull, 131072ull};
  bf16* wm[6];
  for (int i = 0; i < 6; ++i) wm[i] = (bf16*)alloc(wElems[i] * 2);
  float* sums = (float*)alloc(1024 * 4);
  float* scsh = (float*)alloc(1024 * 4);

  // Layer configs
  const int C_in[6]  = {2, 64, 128, 256, 512, 512};
  const int C_out[6] = {64, 128, 256, 512, 512, 1};
  const int C_pad[6] = {64, 128, 256, 512, 512, 16};
  const int cinsh[6] = {1, 6, 7, 8, 9, 9};
  const int Hin[6]   = {256, 128, 64, 32, 16, 16};
  const int Hout[6]  = {128, 64, 32, 16, 16, 16};
  const int strd[6]  = {2, 2, 2, 2, 1, 1};
  const int hwsh[6]  = {7, 6, 5, 4, 4, 4};

  // Convert weights to bf16 GEMM matrices
  for (int l = 0; l < 6; ++l)
    prep_weights_kernel<<<dim3(1024), dim3(256), 0, stream>>>(
        w[l], wm[l], C_in[l], C_out[l], C_pad[l], cinsh[l]);

  // Build conv-0 input (channel 0 copy + rasterized endpoint channel)
  prep_input_kernel<<<dim3((64 * 65536 + 255) / 256), dim3(256), 0, stream>>>(x, act[0]);
  rasterize_kernel<<<dim3(64), dim3(256), 0, stream>>>(x, act[0]);

  for (int l = 0; l < 6; ++l) {
    int total_pix = 64 * Hout[l] * Hout[l];
    dim3 grid((total_pix + 255) / 256, C_pad[l] / 16);
    int flags = (l == 0 ? 1 : 0) | (l == 5 ? 2 : 0);
    bf16*  obf  = (l < 5) ? act[l + 1] : nullptr;
    float* of32 = (l == 5) ? (float*)d_out : nullptr;
    conv_wmma_kernel<<<grid, dim3(128), 0, stream>>>(
        act[l], wm[l], bb[l], obf, of32,
        C_in[l], cinsh[l], C_out[l], Hin[l], Hin[l], Hout[l], hwsh[l],
        strd[l], total_pix, flags);
    if (l >= 1 && l <= 4) {
      int C = C_out[l];
      size_t total8 = ((size_t)total_pix * (size_t)C) / 8;
      zero_kernel<<<dim3((2 * C + 255) / 256), dim3(256), 0, stream>>>(sums, 2 * C);
      stats_kernel<<<dim3(2048), dim3(256), 0, stream>>>(act[l + 1], sums, C, total8);
      bn_finalize_kernel<<<dim3((C + 255) / 256), dim3(256), 0, stream>>>(
          sums, g[l], be[l], scsh, C, 1.0f / (float)total_pix);
      bn_apply_kernel<<<dim3(2048), dim3(256), 0, stream>>>(act[l + 1], scsh, C, total8);
    }
  }
}
